// SymptomRoutedRGAT_22136261444406
// MI455X (gfx1250) — compile-verified, matrix operand order checked by reading, not run on hardware
//
#include <hip/hip_runtime.h>
#include <hip/hip_bf16.h>
#include <math.h>

// ---------------------------------------------------------------------------
// SymptomRoutedRGAT for MI455X (gfx1250, wave32, WMMA).
// All GEMM-shaped work runs through v_wmma_f32_16x16x32_f16 (f16 in, f32 acc).
// Edges are bucketed by relation type so 16-edge tiles share one weight matrix
// and the irregular gather becomes an LDS-staged WMMA GEMM. Row gathers use the
// gfx1250 async global->LDS path (global_load_async_to_lds_b128 + ASYNCcnt).
// ---------------------------------------------------------------------------

typedef _Float16 f16;
typedef __attribute__((ext_vector_type(16))) _Float16 v16h;
typedef __attribute__((ext_vector_type(8)))  _Float16 h8;
typedef __attribute__((ext_vector_type(8)))  float    v8f;

#define DEV static __device__ __forceinline__

constexpr int NN  = 50000;
constexpr int EE  = 100000;
constexpr int HH  = 256;
constexpr int RR  = 6;
constexpr int KK  = 8;
constexpr int NHH = 4;
constexpr int HDD = 64;
constexpr int NTILES = NN / 16;           // 3125 (exact)
constexpr int ETILES = EE / 16;           // 6250 (exact)

// meta[] int layout in workspace
constexpr int META_CNT = 0;   // [0..5]  per-type edge counts
constexpr int META_OFF = 8;   // [8..13] per-type bucket offsets
constexpr int META_CUR = 16;  // [16..21] per-type scatter cursors

DEV v16h make16(h8 lo, h8 hi) {
  return __builtin_shufflevector(lo, hi, 0,1,2,3,4,5,6,7,8,9,10,11,12,13,14,15);
}

// A fragment (16x32 f16), row-major source with leading dim lda (halfs).
// Lane L: row = L&15, koff = (L>>4)*8; elements {koff..koff+7, 16+koff..16+koff+7}.
DEV v16h load_a(const f16* A, int lda, int lane, int k0) {
  const int r = lane & 15, koff = (lane >> 4) << 3;
  const f16* p = A + (size_t)r * lda + k0 + koff;
  h8 lo = *(const h8*)p;
  h8 hi = *(const h8*)(p + 16);
  return make16(lo, hi);
}

// B fragment (32x16 f16 = KxN). W stored (out,in) row-major, ldw = K-stride in halfs,
// so per lane the 8-element groups are contiguous 16B loads.
DEV v16h load_b(const f16* W, int ldw, int colg, int lane, int k0) {
  const int c = lane & 15, koff = (lane >> 4) << 3;
  const f16* p = W + (size_t)(colg + c) * ldw + k0 + koff;
  h8 lo = *(const h8*)p;
  h8 hi = *(const h8*)(p + 16);
  return make16(lo, hi);
}

DEV v8f wmma(v16h a, v16h b, v8f c) {
  return __builtin_amdgcn_wmma_f32_16x16x32_f16(false, a, false, b, (short)0, c, false, false);
}

// gfx1250 async global->LDS copy: 32 bytes per lane (two b128, shared addr regs;
// INST_OFFSET applies to both LDS and global addresses). Tracked by ASYNCcnt.
DEV void async_gather32(f16* lds_dst, const f16* gsrc) {
  unsigned l = (unsigned)(uintptr_t)lds_dst;   // low 32 bits = LDS byte offset
  asm volatile("global_load_async_to_lds_b128 %0, %1, off\n\t"
               "global_load_async_to_lds_b128 %0, %1, off offset:16"
               :: "v"(l), "v"(gsrc) : "memory");
}
DEV void wait_async0() {
  asm volatile("s_wait_asynccnt 0x0" ::: "memory");
}

// order-preserving float<->uint encode for atomicMax on signed floats
DEV unsigned enc_f(float v) {
  int b = __float_as_int(v);
  return (b < 0) ? (unsigned)~b : ((unsigned)b | 0x80000000u);
}
DEV float dec_f(unsigned k) {
  return (k & 0x80000000u) ? __int_as_float((int)(k ^ 0x80000000u))
                           : __int_as_float(~(int)k);
}

// ---------------------------------------------------------------------------
// casts
// ---------------------------------------------------------------------------
__global__ void k_cast(const float* __restrict__ s, f16* __restrict__ d, size_t n) {
  size_t st = (size_t)gridDim.x * blockDim.x;
  for (size_t i = (size_t)blockIdx.x * blockDim.x + threadIdx.x; i < n; i += st)
    d[i] = (f16)s[i];
}

// (rows,cols) row-major f32 -> transposed (cols,rows) f16  ((in,out) -> (out,in))
__global__ void k_tcast(const float* __restrict__ s, f16* __restrict__ d, int rows, int cols) {
  size_t n = (size_t)rows * cols;
  size_t st = (size_t)gridDim.x * blockDim.x;
  for (size_t i = (size_t)blockIdx.x * blockDim.x + threadIdx.x; i < n; i += st) {
    int r = (int)(i / cols), c = (int)(i % cols);
    d[(size_t)c * rows + r] = (f16)s[i];
  }
}

// ---------------------------------------------------------------------------
// edge bucketing by type
// ---------------------------------------------------------------------------
__global__ void k_hist(const int* __restrict__ et, int* __restrict__ meta) {
  int st = gridDim.x * blockDim.x;
  for (int e = blockIdx.x * blockDim.x + threadIdx.x; e < EE; e += st)
    atomicAdd(&meta[META_CNT + et[e]], 1);
}
__global__ void k_offs(int* meta) {
  if (threadIdx.x == 0) {
    int off = 0;
    for (int t = 0; t < RR; ++t) { meta[META_OFF + t] = off; off += meta[META_CNT + t]; }
  }
}
__global__ void k_bucket(const int* __restrict__ et, int* __restrict__ meta,
                         int* __restrict__ ebuf) {
  int st = gridDim.x * blockDim.x;
  for (int e = blockIdx.x * blockDim.x + threadIdx.x; e < EE; e += st) {
    int t = et[e];
    int pos = meta[META_OFF + t] + atomicAdd(&meta[META_CUR + t], 1);
    ebuf[pos] = e;
  }
}

// ---------------------------------------------------------------------------
// routing MLP layer 1: x1 = relu(h @ mlp_w1 + b1)   (WMMA, f16 out)
// ---------------------------------------------------------------------------
__global__ __launch_bounds__(256) void k_x1(const f16* __restrict__ hh,
                                            const f16* __restrict__ w1t,
                                            const float* __restrict__ b1,
                                            f16* __restrict__ x1) {
  const int n0 = blockIdx.x * 16;
  const int wave = threadIdx.x >> 5, lane = threadIdx.x & 31;
  const int colg = blockIdx.y * 128 + wave * 16;
  v8f acc = {0.f,0.f,0.f,0.f,0.f,0.f,0.f,0.f};
  const f16* A = hh + (size_t)n0 * HH;
#pragma unroll
  for (int k0 = 0; k0 < HH; k0 += 32) {
    v16h a = load_a(A, HH, lane, k0);
    v16h b = load_b(w1t, HH, colg, lane, k0);
    acc = wmma(a, b, acc);
  }
  const int col = colg + (lane & 15), rb = (lane >> 4) << 3;
  const float bv = b1[col];
#pragma unroll
  for (int i = 0; i < 8; ++i) {
    float v = fmaxf(acc[i] + bv, 0.f);
    x1[(size_t)(n0 + rb + i) * HH + col] = (f16)v;
  }
}

// routing MLP layer 2 + softmax: gate_vals[n,k]  (one wave per node)
__global__ __launch_bounds__(256) void k_gate(const f16* __restrict__ x1,
                                              const float* __restrict__ w2,
                                              const float* __restrict__ b2,
                                              float* __restrict__ gv) {
  const int lane = threadIdx.x & 31;
  const int node = blockIdx.x * (blockDim.x >> 5) + (threadIdx.x >> 5);
  if (node >= NN) return;
  float p[KK] = {};
  for (int i = lane; i < HH; i += 32) {
    float x = (float)x1[(size_t)node * HH + i];
#pragma unroll
    for (int k = 0; k < KK; ++k) p[k] += x * w2[i * KK + k];
  }
#pragma unroll
  for (int k = 0; k < KK; ++k)
    for (int m = 16; m; m >>= 1) p[k] += __shfl_xor(p[k], m, 32);
  if (lane == 0) {
    float mx = -1e30f;
#pragma unroll
    for (int k = 0; k < KK; ++k) { p[k] += b2[k]; mx = fmaxf(mx, p[k]); }
    float s = 0.f;
#pragma unroll
    for (int k = 0; k < KK; ++k) { p[k] = __expf(p[k] - mx); s += p[k]; }
    float inv = 1.f / s;
#pragma unroll
    for (int k = 0; k < KK; ++k) gv[(size_t)node * KK + k] = p[k] * inv;
  }
}

// sym_w = softmax(sym_edge_logits, axis=1)
__global__ void k_symw(const float* __restrict__ logits, float* __restrict__ sw) {
  int k = threadIdx.x;
  if (k >= KK) return;
  float mx = -1e30f;
  for (int r = 0; r < RR; ++r) mx = fmaxf(mx, logits[k * RR + r]);
  float e[RR], s = 0.f;
  for (int r = 0; r < RR; ++r) { e[r] = __expf(logits[k * RR + r] - mx); s += e[r]; }
  for (int r = 0; r < RR; ++r) sw[k * RR + r] = e[r] / s;
}

// ---------------------------------------------------------------------------
// attention scores: per 16-edge tile of one type, q/k via WMMA gather-GEMM
// ---------------------------------------------------------------------------
__global__ __launch_bounds__(256) void k_scores(const f16* __restrict__ hh,
                                                const f16* __restrict__ wq,
                                                const f16* __restrict__ wk,
                                                const int* __restrict__ ei,
                                                const int* __restrict__ ebuf,
                                                const int* __restrict__ meta,
                                                float* __restrict__ esc) {
  const int t = blockIdx.y;
  const int cnt = meta[META_CNT + t];
  const int start = blockIdx.x * 16;
  if (start >= cnt) return;
  const int base = meta[META_OFF + t];

  __shared__ __align__(16) f16 Adst[16 * HH];
  __shared__ __align__(16) f16 Asrc[16 * HH];
  __shared__ float Qs[16 * HH];
  __shared__ float Ks[16 * HH];
  __shared__ int sdst[16], ssrc[16];

  const int tid = threadIdx.x;
  const f16* wqb = wq + (size_t)t * HH * HH;
  const f16* wkb = wk + (size_t)t * HH * HH;
  __builtin_prefetch(wqb, 0, 0);            // global_prefetch_b8 toward this WGP
  __builtin_prefetch(wkb, 0, 0);

  if (tid < 16) {
    int idx = start + tid;
    int e = ebuf[base + ((idx < cnt) ? idx : 0)];
    sdst[tid] = ei[EE + e];
    ssrc[tid] = ei[e];
  }
  __syncthreads();
  {
    // async global->LDS gather: 32B per thread per matrix, no VGPR round trip
    const int row = tid >> 4, seg = tid & 15;
    async_gather32(Adst + row * HH + seg * 16, hh + (size_t)sdst[row] * HH + seg * 16);
    async_gather32(Asrc + row * HH + seg * 16, hh + (size_t)ssrc[row] * HH + seg * 16);
    wait_async0();
  }
  __syncthreads();

  const int wave = tid >> 5, lane = tid & 31;
  for (int u = wave; u < 32; u += 8) {          // 32 tiles: 16 for Q, 16 for K
    const int mat = u >> 4, colg = (u & 15) * 16;
    const f16* A = mat ? Asrc : Adst;
    const f16* W = mat ? wkb : wqb;
    v8f acc = {0.f,0.f,0.f,0.f,0.f,0.f,0.f,0.f};
#pragma unroll
    for (int k0 = 0; k0 < HH; k0 += 32) {
      v16h a = load_a(A, HH, lane, k0);
      v16h b = load_b(W, HH, colg, lane, k0);
      acc = wmma(a, b, acc);
    }
    float* O = mat ? Ks : Qs;
    const int col = colg + (lane & 15), rb = (lane >> 4) << 3;
#pragma unroll
    for (int i = 0; i < 8; ++i) O[(rb + i) * HH + col] = acc[i];
  }
  __syncthreads();

  if (tid < 64) {
    int e = tid >> 2, head = tid & 3;
    if (start + e < cnt) {
      const float* q = Qs + e * HH + head * HDD;
      const float* kk = Ks + e * HH + head * HDD;
      float s = 0.f;
#pragma unroll 8
      for (int d = 0; d < HDD; ++d) s += q[d] * kk[d];
      esc[(size_t)(base + start + e) * NHH + head] = s * 0.125f;  // /sqrt(64)
    }
  }
}

// scatter-softmax pass 1: segment max (encoded-uint atomicMax)
__global__ void k_smax(const int* __restrict__ ei, const int* __restrict__ et,
                       const int* __restrict__ ebuf, const float* __restrict__ esc,
                       unsigned* __restrict__ mbuf) {
  int st = gridDim.x * blockDim.x;
  for (int p = blockIdx.x * blockDim.x + threadIdx.x; p < EE; p += st) {
    int e = ebuf[p], t = et[e], d = ei[EE + e];
    size_t bi = ((size_t)d * RR + t) * NHH;
#pragma unroll
    for (int h = 0; h < NHH; ++h)
      atomicMax(&mbuf[bi + h], enc_f(esc[(size_t)p * NHH + h]));
  }
}

// scatter-softmax pass 2: exp + segment sum
__global__ void k_sexp(const int* __restrict__ ei, const int* __restrict__ et,
                       const int* __restrict__ ebuf, float* __restrict__ esc,
                       const unsigned* __restrict__ mbuf, float* __restrict__ dbuf) {
  int st = gridDim.x * blockDim.x;
  for (int p = blockIdx.x * blockDim.x + threadIdx.x; p < EE; p += st) {
    int e = ebuf[p], t = et[e], d = ei[EE + e];
    size_t bi = ((size_t)d * RR + t) * NHH;
#pragma unroll
    for (int h = 0; h < NHH; ++h) {
      size_t pi = (size_t)p * NHH + h;
      float ex = __expf(esc[pi] - dec_f(mbuf[bi + h]));
      esc[pi] = ex;
      atomicAdd(&dbuf[bi + h], ex);
    }
  }
}

// ---------------------------------------------------------------------------
// aggregation: V via WMMA gather-GEMM, alpha-weighted scatter-add into agg
// ---------------------------------------------------------------------------
__global__ __launch_bounds__(256) void k_agg(const f16* __restrict__ hh,
                                             const f16* __restrict__ wv,
                                             const int* __restrict__ ei,
                                             const int* __restrict__ ebuf,
                                             const int* __restrict__ meta,
                                             const float* __restrict__ esc,
                                             const float* __restrict__ dbuf,
                                             float* __restrict__ agg) {
  const int t = blockIdx.y;
  const int cnt = meta[META_CNT + t];
  const int start = blockIdx.x * 16;
  if (start >= cnt) return;
  const int base = meta[META_OFF + t];

  __shared__ __align__(16) f16 Asrc[16 * HH];
  __shared__ float Vs[16 * HH];
  __shared__ int sdst[16], ssrc[16];
  __shared__ float alph[16][NHH];

  const int tid = threadIdx.x;
  const f16* wvb = wv + (size_t)t * HH * HH;
  __builtin_prefetch(wvb, 0, 0);

  if (tid < 16) {
    int idx = start + tid;
    int e = ebuf[base + ((idx < cnt) ? idx : 0)];
    sdst[tid] = ei[EE + e];
    ssrc[tid] = ei[e];
  }
  __syncthreads();
  {
    const int row = tid >> 4, seg = tid & 15;
    async_gather32(Asrc + row * HH + seg * 16, hh + (size_t)ssrc[row] * HH + seg * 16);
    wait_async0();
  }
  __syncthreads();

  const int wave = tid >> 5, lane = tid & 31;
  for (int u = wave * 2; u < wave * 2 + 2; ++u) {
    const int colg = u * 16;
    v8f acc = {0.f,0.f,0.f,0.f,0.f,0.f,0.f,0.f};
#pragma unroll
    for (int k0 = 0; k0 < HH; k0 += 32) {
      v16h a = load_a(Asrc, HH, lane, k0);
      v16h b = load_b(wvb, HH, colg, lane, k0);
      acc = wmma(a, b, acc);
    }
    const int col = colg + (lane & 15), rb = (lane >> 4) << 3;
#pragma unroll
    for (int i = 0; i < 8; ++i) Vs[(rb + i) * HH + col] = acc[i];
  }
  __syncthreads();

  if (tid < 64) {
    int e = tid >> 2, head = tid & 3;
    float a = 0.f;
    if (start + e < cnt) {
      size_t p = (size_t)(base + start + e) * NHH + head;
      float den = dbuf[((size_t)sdst[e] * RR + t) * NHH + head];
      a = esc[p] / fmaxf(den, 1e-16f);
    }
    alph[e][head] = a;
  }
  __syncthreads();

  {
    const int row = tid >> 4, seg = tid & 15;
    if (start + row < cnt) {
      size_t bi = ((size_t)sdst[row] * RR + t) * HH;
#pragma unroll
      for (int j = 0; j < 16; ++j) {
        int c = seg * 16 + j;
        atomicAdd(&agg[bi + c], Vs[row * HH + c] * alph[row][c >> 6]);
      }
    }
  }
}

// ---------------------------------------------------------------------------
// fused symptom path: mix over R -> relu(W_sym)·gate -> W_cross slice (atomic acc)
// ---------------------------------------------------------------------------
__global__ __launch_bounds__(256) void k_sym(const float* __restrict__ agg,
                                             const float* __restrict__ sw,
                                             const float* __restrict__ gv,
                                             const f16* __restrict__ wsym,
                                             const float* __restrict__ bsym,
                                             const f16* __restrict__ wcT,
                                             float* __restrict__ cacc) {
  const int n0 = blockIdx.x * 16, k = blockIdx.y;
  __shared__ __align__(16) f16 Sh[16 * HH];
  __shared__ __align__(16) f16 Oh[16 * HH];
  __shared__ float grow[16];

  float swk[RR];
#pragma unroll
  for (int r = 0; r < RR; ++r) swk[r] = sw[k * RR + r];

  const int tid = threadIdx.x, row = tid >> 4, seg = tid & 15;
  if (tid < 16) grow[tid] = gv[(size_t)(n0 + tid) * KK + k];
  {
    const float* ab = agg + (size_t)(n0 + row) * RR * HH;
#pragma unroll
    for (int j = 0; j < 16; ++j) {
      int c = seg * 16 + j;
      float a = 0.f;
#pragma unroll
      for (int r = 0; r < RR; ++r) a += swk[r] * ab[r * HH + c];
      Sh[row * HH + c] = (f16)a;
    }
  }
  __syncthreads();

  const int wave = tid >> 5, lane = tid & 31;
  const f16* wb = wsym + (size_t)k * HH * HH;
  for (int u = wave * 2; u < wave * 2 + 2; ++u) {
    const int colg = u * 16;
    v8f acc = {0.f,0.f,0.f,0.f,0.f,0.f,0.f,0.f};
#pragma unroll
    for (int k0 = 0; k0 < HH; k0 += 32) {
      v16h a = load_a(Sh, HH, lane, k0);
      v16h b = load_b(wb, HH, colg, lane, k0);
      acc = wmma(a, b, acc);
    }
    const int col = colg + (lane & 15), rb = (lane >> 4) << 3;
    const float bv = bsym[k * HH + col];
#pragma unroll
    for (int i = 0; i < 8; ++i) {
      float v = fmaxf(acc[i] + bv, 0.f) * grow[rb + i];
      Oh[(rb + i) * HH + col] = (f16)v;
    }
  }
  __syncthreads();

  for (int u = wave * 2; u < wave * 2 + 2; ++u) {
    const int colg = u * 16;
    v8f acc = {0.f,0.f,0.f,0.f,0.f,0.f,0.f,0.f};
#pragma unroll
    for (int k0 = 0; k0 < HH; k0 += 32) {
      v16h a = load_a(Oh, HH, lane, k0);
      v16h b = load_b(wcT, KK * HH, colg, lane, k * HH + k0);  // W_cross^T rows [k*H, k*H+256)
      acc = wmma(a, b, acc);
    }
    const int col = colg + (lane & 15), rb = (lane >> 4) << 3;
#pragma unroll
    for (int i = 0; i < 8; ++i)
      atomicAdd(&cacc[(size_t)(n0 + rb + i) * HH + col], acc[i]);
  }
}

// ---------------------------------------------------------------------------
// final: h_new = relu(cacc+b_cross); gate = sigmoid([h,h_new]@Wg+bg); layernorm
// ---------------------------------------------------------------------------
__global__ __launch_bounds__(256) void k_final(const float* __restrict__ h,
                                               const float* __restrict__ cacc,
                                               const float* __restrict__ bc,
                                               const f16* __restrict__ wgT,
                                               const float* __restrict__ bg,
                                               const float* __restrict__ lng,
                                               const float* __restrict__ lnb,
                                               float* __restrict__ out) {
  const int n0 = blockIdx.x * 16;
  __shared__ __align__(16) f16 Acat[16 * 2 * HH];
  __shared__ float HN[16 * HH];
  __shared__ float Y[16 * HH];

  const int tid = threadIdx.x, row = tid >> 4, seg = tid & 15;
  {
    size_t gb = (size_t)(n0 + row) * HH;
#pragma unroll
    for (int j = 0; j < 16; ++j) {
      int c = seg * 16 + j;
      float hv = h[gb + c];
      Acat[row * (2 * HH) + c] = (f16)hv;
      float hn = fmaxf(cacc[gb + c] + bc[c], 0.f);
      HN[row * HH + c] = hn;
      Acat[row * (2 * HH) + HH + c] = (f16)hn;
    }
  }
  __syncthreads();

  const int wave = tid >> 5, lane = tid & 31;
  for (int u = wave * 2; u < wave * 2 + 2; ++u) {
    const int colg = u * 16;
    v8f acc = {0.f,0.f,0.f,0.f,0.f,0.f,0.f,0.f};
#pragma unroll
    for (int k0 = 0; k0 < 2 * HH; k0 += 32) {
      v16h a = load_a(Acat, 2 * HH, lane, k0);
      v16h b = load_b(wgT, 2 * HH, colg, lane, k0);
      acc = wmma(a, b, acc);
    }
    const int col = colg + (lane & 15), rb = (lane >> 4) << 3;
    const float bgv = bg[col];
#pragma unroll
    for (int i = 0; i < 8; ++i) {
      int rr = rb + i;
      float g = 1.f / (1.f + __expf(-(acc[i] + bgv)));
      float hv = h[(size_t)(n0 + rr) * HH + col];
      Y[rr * HH + col] = g * HN[rr * HH + col] + (1.f - g) * hv;
    }
  }
  __syncthreads();

  for (int rr = wave * 2; rr < wave * 2 + 2; ++rr) {
    float s = 0.f, ss = 0.f;
    for (int c = lane; c < HH; c += 32) {
      float y = Y[rr * HH + c];
      s += y; ss += y * y;
    }
    for (int m = 16; m; m >>= 1) {
      s += __shfl_xor(s, m, 32);
      ss += __shfl_xor(ss, m, 32);
    }
    const float mu = s * (1.f / HH);
    const float var = ss * (1.f / HH) - mu * mu;
    const float rstd = rsqrtf(var + 1e-5f);
    for (int c = lane; c < HH; c += 32)
      out[(size_t)(n0 + rr) * HH + c] = lng[c] * (Y[rr * HH + c] - mu) * rstd + lnb[c];
  }
}

// ---------------------------------------------------------------------------
// host launcher
// ---------------------------------------------------------------------------
extern "C" void kernel_launch(void* const* d_in, const int* in_sizes, int n_in,
                              void* d_out, int out_size, void* d_ws, size_t ws_size,
                              hipStream_t stream) {
  const float* h    = (const float*)d_in[0];
  const int*   ei   = (const int*)  d_in[1];
  const int*   et   = (const int*)  d_in[2];
  const float* Wq   = (const float*)d_in[3];
  const float* Wk   = (const float*)d_in[4];
  const float* Wv   = (const float*)d_in[5];
  const float* w1   = (const float*)d_in[6];
  const float* b1   = (const float*)d_in[7];
  const float* w2   = (const float*)d_in[8];
  const float* b2   = (const float*)d_in[9];
  const float* slog = (const float*)d_in[10];
  const float* Wsym = (const float*)d_in[11];
  const float* bsym = (const float*)d_in[12];
  const float* Wc   = (const float*)d_in[13];
  const float* bc   = (const float*)d_in[14];
  const float* Wg   = (const float*)d_in[15];
  const float* bg   = (const float*)d_in[16];
  const float* lng  = (const float*)d_in[17];
  const float* lnb  = (const float*)d_in[18];
  float* out = (float*)d_out;

  char* ws = (char*)d_ws;
  size_t off = 0;
  auto alloc = [&](size_t bytes) -> char* {
    char* p = ws + off;
    off = (off + bytes + 255) & ~(size_t)255;
    return p;
  };

  f16* h_h    = (f16*)alloc((size_t)NN * HH * 2);
  f16* x1     = (f16*)alloc((size_t)NN * HH * 2);
  f16* wq_h   = (f16*)alloc((size_t)RR * HH * HH * 2);
  f16* wk_h   = (f16*)alloc((size_t)RR * HH * HH * 2);
  f16* wv_h   = (f16*)alloc((size_t)RR * HH * HH * 2);
  f16* wsym_h = (f16*)alloc((size_t)KK * HH * HH * 2);
  f16* wcT    = (f16*)alloc((size_t)KK * HH * HH * 2);   // (256, 2048)
  f16* wgT    = (f16*)alloc((size_t)2 * HH * HH * 2);    // (256, 512)
  f16* w1t    = (f16*)alloc((size_t)HH * HH * 2);        // (256, 256)
  float* gv   = (float*)alloc((size_t)NN * KK * 4);
  float* sw   = (float*)alloc(256);
  int* meta   = (int*)alloc(64 * 4);
  int* ebuf   = (int*)alloc((size_t)EE * 4);
  float* esc  = (float*)alloc((size_t)EE * NHH * 4);
  unsigned* mbuf = (unsigned*)alloc((size_t)NN * RR * NHH * 4);
  float* dbuf = (float*)alloc((size_t)NN * RR * NHH * 4);
  float* agg  = (float*)alloc((size_t)NN * RR * HH * 4);
  float* cacc = (float*)alloc((size_t)NN * HH * 4);
  (void)ws_size; (void)in_sizes; (void)n_in; (void)out_size;

  // zero the accumulators / counters (mbuf=0 is below enc_f of any non-NaN float)
  hipMemsetAsync(meta, 0, 64 * 4, stream);
  hipMemsetAsync(mbuf, 0, (size_t)NN * RR * NHH * 4, stream);
  hipMemsetAsync(dbuf, 0, (size_t)NN * RR * NHH * 4, stream);
  hipMemsetAsync(agg,  0, (size_t)NN * RR * HH * 4, stream);
  hipMemsetAsync(cacc, 0, (size_t)NN * HH * 4, stream);

  // f16 conversions
  k_cast<<<2048, 256, 0, stream>>>(h, h_h, (size_t)NN * HH);
  k_cast<<<512, 256, 0, stream>>>(Wq, wq_h, (size_t)RR * HH * HH);
  k_cast<<<512, 256, 0, stream>>>(Wk, wk_h, (size_t)RR * HH * HH);
  k_cast<<<512, 256, 0, stream>>>(Wv, wv_h, (size_t)RR * HH * HH);
  k_cast<<<512, 256, 0, stream>>>(Wsym, wsym_h, (size_t)KK * HH * HH);
  k_tcast<<<512, 256, 0, stream>>>(w1, w1t, HH, HH);
  k_tcast<<<512, 256, 0, stream>>>(Wc, wcT, KK * HH, HH);
  k_tcast<<<256, 256, 0, stream>>>(Wg, wgT, 2 * HH, HH);

  // edge bucketing
  k_hist<<<256, 256, 0, stream>>>(et, meta);
  k_offs<<<1, 32, 0, stream>>>(meta);
  k_bucket<<<256, 256, 0, stream>>>(et, meta, ebuf);

  // routing MLP + symptom-edge softmax
  k_x1<<<dim3(NTILES, 2), 256, 0, stream>>>(h_h, w1t, b1, x1);
  k_gate<<<(NN + 7) / 8, 256, 0, stream>>>(x1, w2, b2, gv);
  k_symw<<<1, 32, 0, stream>>>(slog, sw);

  // per-type attention
  k_scores<<<dim3(ETILES, RR), 256, 0, stream>>>(h_h, wq_h, wk_h, ei, ebuf, meta, esc);
  k_smax<<<512, 256, 0, stream>>>(ei, et, ebuf, esc, mbuf);
  k_sexp<<<512, 256, 0, stream>>>(ei, et, ebuf, esc, mbuf, dbuf);
  k_agg<<<dim3(ETILES, RR), 256, 0, stream>>>(h_h, wv_h, ei, ebuf, meta, esc, dbuf, agg);

  // symptom mixing + W_sym + W_cross (fused, atomic accumulation)
  k_sym<<<dim3(NTILES, KK), 256, 0, stream>>>(agg, sw, gv, wsym_h, bsym, wcT, cacc);

  // gate + residual + layernorm
  k_final<<<NTILES, 256, 0, stream>>>(h, cacc, bc, wgT, bg, lng, lnb, out);
}
